// MultiHeadAttention_46316927320223
// MI455X (gfx1250) — compile-verified, matrix operand order checked by reading, not run on hardware
//
#include <hip/hip_runtime.h>
#include <hip/hip_bf16.h>
#include <math.h>

// ---------------------------------------------------------------------------
// MI455X (gfx1250) fused multi-head attention, bf16 WMMA path.
//  B=2, S=2048, D=1024, H=16, DK=64.  ~103 GFLOP vs ~100MB HBM traffic ->
//  compute bound on the matrix pipes; all four GEMM stages run on
//  v_wmma_f32_16x16x32_bf16.  Attention K/V tiles are staged into LDS by the
//  Tensor Data Mover (TENSORcnt) when available, else async copies (ASYNCcnt).
// ---------------------------------------------------------------------------

#define Bc 2
#define Sc 2048
#define Dm 1024
#define Hc 16
#define DKc 64

typedef __attribute__((ext_vector_type(16))) __bf16 v16bf;
typedef __attribute__((ext_vector_type(8)))  __bf16 v8bf;
typedef __attribute__((ext_vector_type(8)))  float  v8f;
typedef int v4i __attribute__((vector_size(16)));
typedef unsigned int u32x4 __attribute__((vector_size(16)));
typedef int i32x8 __attribute__((vector_size(32)));
typedef int i32x4 __attribute__((vector_size(16)));

#if defined(__has_builtin)
#if __has_builtin(__builtin_amdgcn_tensor_load_to_lds) && \
    __has_builtin(__builtin_amdgcn_s_wait_tensorcnt)
#define USE_TDM 1
#endif
#if __has_builtin(__builtin_amdgcn_global_load_async_to_lds_b128) && \
    __has_builtin(__builtin_amdgcn_s_wait_asynccnt)
#define USE_ASYNC 1
#endif
#endif
#ifndef USE_TDM
#define USE_TDM 0
#endif
#ifndef USE_ASYNC
#define USE_ASYNC 0
#endif

#if USE_ASYNC
typedef __attribute__((address_space(1))) v4i* gv4i_p;   // global (AS1)
typedef __attribute__((address_space(3))) v4i* lv4i_p;   // LDS    (AS3)
#endif

union FragU { v16bf v; v8bf h[2]; };

// A-fragment (16x32 bf16, ISA 7.12.2): lane&15 = M row; lanes>=16 take K+8.
// B operand uses the same loader applied to B^T (row = N, k = K).
__device__ __forceinline__ v16bf load_frag_bf16(const __bf16* base, int ld, int lane) {
    const int row = lane & 15;
    const int kb  = (lane >> 4) << 3;            // 0 or 8
    const __bf16* p = base + (size_t)row * ld;
    FragU u;
    u.h[0] = *(const v8bf*)(p + kb);             // K = kb .. kb+7
    u.h[1] = *(const v8bf*)(p + 16 + kb);        // K = 16+kb .. 16+kb+7
    return u.v;
}

__device__ __forceinline__ v16bf load_frag_f32(const float* base, int ld, int lane) {
    const int row = lane & 15;
    const int kb  = (lane >> 4) << 3;
    const float* p = base + (size_t)row * ld;
    FragU u;
#pragma unroll
    for (int j = 0; j < 8; ++j) {
        u.v[j]     = (__bf16)p[kb + j];
        u.v[8 + j] = (__bf16)p[16 + kb + j];
    }
    return u.v;
}

__device__ __forceinline__ v8f wmma_bf16(v16bf a, v16bf b, v8f c) {
    return __builtin_amdgcn_wmma_f32_16x16x32_bf16(
        /*neg_a=*/false, a, /*neg_b=*/false, b,
        /*c_mod=*/(short)0, c, /*reuse_a=*/false, /*reuse_b=*/false);
}

// 16-byte global -> LDS copy: async (ASYNCcnt) when available, else via regs.
__device__ __forceinline__ void cp16_g2l(const __bf16* g, __bf16* l) {
#if USE_ASYNC
    __builtin_amdgcn_global_load_async_to_lds_b128(
        (gv4i_p)(v4i*)g, (lv4i_p)(v4i*)l, 0, 0);
#else
    *(v8bf*)l = *(const v8bf*)g;
#endif
}

#if USE_TDM
// 32-bit LDS offset of a __shared__ object (ptrtoint of AS3 pointer).
__device__ __forceinline__ unsigned lds_off(const void* p) {
    return (unsigned)(unsigned long long)
        (__attribute__((address_space(3))) void*)(void*)p;
}

// TDM 2D tile load, bf16 elements (data_size=1 -> 2 bytes).
// D# per CDNA5 ISA: group0 = {count=1, lds_addr,
// global_addr[56:0], type=2}; group1 = dims/strides; groups 2-3 zero (2D).
// This toolchain exposes the 6-arg builtin (extra i32x8 before cpol).
__device__ __forceinline__ void tdm_load_2d(unsigned ldsA, const void* gaddr,
                                            unsigned tw, unsigned th,
                                            unsigned stride) {
    const unsigned long long ga = (unsigned long long)gaddr;
    u32x4 g0;
    g0[0] = 1u;                                        // count=1, user D#
    g0[1] = ldsA;                                      // lds_addr (bytes)
    g0[2] = (unsigned)(ga & 0xffffffffu);              // global_addr[31:0]
    g0[3] = (unsigned)((ga >> 32) & 0x01ffffffu)       // global_addr[56:32]
          | (2u << 30);                                // type = 2 ("image")
    unsigned w0[8];
    w0[0] = (1u << 16);                                // data_size=1 (2B)
    w0[1] = (tw & 0xffffu) << 16;                      // tensor_dim0[15:0]
    w0[2] = (tw >> 16) | ((th & 0xffffu) << 16);       // dim0 hi | dim1 lo
    w0[3] = (th >> 16) | (tw << 16);                   // dim1 hi | tile_dim0
    w0[4] = th;                                        // tile_dim1 (tile_dim2=0)
    w0[5] = stride;                                    // tensor_dim0_stride lo
    w0[6] = 0u;                                        // stride hi | dim1_stride lo
    w0[7] = 0u;
    i32x8 g1;
#pragma unroll
    for (int i = 0; i < 8; ++i) g1[i] = (int)w0[i];
    i32x4 z4 = {0, 0, 0, 0};
    i32x8 z8 = {0, 0, 0, 0, 0, 0, 0, 0};
    __builtin_amdgcn_tensor_load_to_lds(g0, g1, z4, z4, z8, 0);
}
#endif

// ---------------------------------------------------------------------------
// Kernel 0: weight prep.  Wt[n][k] = (bf16) W[k][n]   (B^T layout for WMMA)
// ---------------------------------------------------------------------------
__global__ __launch_bounds__(256) void prep_w(
    const float* __restrict__ Wq, const float* __restrict__ Wk,
    const float* __restrict__ Wv, const float* __restrict__ Wo,
    __bf16* __restrict__ WtQ, __bf16* __restrict__ WtK,
    __bf16* __restrict__ WtV, __bf16* __restrict__ WtO)
{
    const int idx = blockIdx.x * 256 + threadIdx.x;   // over Dm*Dm
    const int z   = blockIdx.y;
    const float* W = (z == 0) ? Wq : (z == 1) ? Wk : (z == 2) ? Wv : Wo;
    __bf16*     Wt = (z == 0) ? WtQ : (z == 1) ? WtK : (z == 2) ? WtV : WtO;
    const int n = idx >> 10;          // output feature
    const int k = idx & (Dm - 1);     // input feature
    Wt[idx] = (__bf16)W[(size_t)k * Dm + n];
}

// ---------------------------------------------------------------------------
// Kernel 1: QKV projection.  Y = X*W + b, head-split outputs:
//   z=0: Q -> Qb [B,H,S,DK] bf16   z=1: K -> Kb [B,H,S,DK] bf16
//   z=2: V -> Vt [B,H,DK,S] bf16 (transposed so PV B-frags are contiguous)
// ---------------------------------------------------------------------------
__global__ __launch_bounds__(256) void qkv_proj(
    const float* __restrict__ Xq, const float* __restrict__ Xk,
    const float* __restrict__ Xv,
    const __bf16* __restrict__ WtQ, const __bf16* __restrict__ WtK,
    const __bf16* __restrict__ WtV,
    const float* __restrict__ bq, const float* __restrict__ bk,
    const float* __restrict__ bv,
    __bf16* __restrict__ Qb, __bf16* __restrict__ Kb, __bf16* __restrict__ Vt)
{
    const int lane = threadIdx.x & 31;
    const int w    = threadIdx.x >> 5;
    const int mt   = blockIdx.x;                 // 0..255 over M = B*S
    const int nt   = blockIdx.y * 8 + w;         // 0..63  over N = D
    const int z    = blockIdx.z;                 // 0=Q 1=K 2=V

    const float*  X    = (z == 0) ? Xq : (z == 1) ? Xk : Xv;
    const __bf16* Wt   = (z == 0) ? WtQ : (z == 1) ? WtK : WtV;
    const float*  bias = (z == 0) ? bq : (z == 1) ? bk : bv;

    const float*  Abase = X  + (size_t)(mt * 16) * Dm;
    const __bf16* Bbase = Wt + (size_t)(nt * 16) * Dm;

    v8f acc = {};
#pragma unroll 4
    for (int kc = 0; kc < Dm; kc += 32) {
        if (kc + 128 < Dm) {                     // WGP-scope global_prefetch_b8
            __builtin_prefetch(Abase + kc + 128 + (size_t)(lane & 15) * Dm, 0, 3);
            __builtin_prefetch(Bbase + kc + 128 + (size_t)(lane & 15) * Dm, 0, 3);
        }
        v16bf a = load_frag_f32 (Abase + kc, Dm, lane);
        v16bf b = load_frag_bf16(Bbase + kc, Dm, lane);
        acc = wmma_bf16(a, b, acc);
    }

    const int half = lane >> 4;
    const int n    = nt * 16 + (lane & 15);
    const int h    = n >> 6;
    const int d    = n & (DKc - 1);
    const float bb = bias[n];
#pragma unroll
    for (int r = 0; r < 8; ++r) {
        const int m  = mt * 16 + r + half * 8;   // global row in [0, B*S)
        const int b_ = m >> 11;                  // / S
        const int s  = m & (Sc - 1);
        const __bf16 yb = (__bf16)(acc[r] + bb);
        if (z == 2) {
            Vt[(((size_t)b_ * Hc + h) * DKc + d) * Sc + s] = yb;
        } else {
            __bf16* dst = (z == 0) ? Qb : Kb;
            dst[(((size_t)b_ * Hc + h) * Sc + s) * DKc + d] = yb;
        }
    }
}

// ---------------------------------------------------------------------------
// Kernel 2: causal flash attention, block-cooperative.
// One block = (b,h) x 128-row q-supertile; 8 waves each own a 16-row q-tile.
// Every 32-key chunk is staged K(32x64) / V(64x32) into LDS, double-buffered:
// TDM descriptors issued by wave 0 (TENSORcnt) when available, else per-lane
// async b128 copies (ASYNCcnt).  All 8 waves consume it (8x L2->LDS reuse).
// Waves skip compute (not barriers) for fully-masked chunks => ~exact causal.
// ---------------------------------------------------------------------------
__global__ __launch_bounds__(256) void attn(
    const __bf16* __restrict__ Qb, const __bf16* __restrict__ Kb,
    const __bf16* __restrict__ Vt, __bf16* __restrict__ Att)
{
    __shared__ __bf16 Ks[2][32][64];             // 8 KB  key-major K chunk
    __shared__ __bf16 Vs[2][64][32];             // 8 KB  dk-major  V chunk
    __shared__ __bf16 Pl[8][16][32];             // 8 KB  per-wave P staging

    const int tid  = threadIdx.x;
    const int lane = tid & 31;
    const int w    = tid >> 5;
    const int bh   = blockIdx.x;                 // 0..31  (B*H)
    const int qs   = blockIdx.y;                 // 0..15  q-supertile (128 rows)
    const int qbase = (qs * 8 + w) * 16;
    const int half = lane >> 4;
    const int nn   = lane & 15;

    const __bf16* Qrow = Qb + ((size_t)bh * Sc + qbase) * DKc;
    const __bf16* Krow = Kb + (size_t)bh * Sc * DKc;
    const __bf16* Vrow = Vt + (size_t)bh * DKc * Sc;

    // cooperative staging coordinates (async fallback: one b128 per thread)
    const int kR = tid >> 3, kC = (tid & 7) * 8;     // K: 32 rows x 64 dk
    const int vR = tid >> 2, vC = (tid & 3) * 8;     // V: 64 rows x 32 keys

    const v16bf aq0 = load_frag_bf16(Qrow,      DKc, lane);  // K dims 0..31
    const v16bf aq1 = load_frag_bf16(Qrow + 32, DKc, lane);  // K dims 32..63

    v8f o0 = {}, o1 = {}, o2 = {}, o3 = {};
    float mrun[8], lrun[8];
#pragma unroll
    for (int r = 0; r < 8; ++r) { mrun[r] = -1.0e30f; lrun[r] = 0.0f; }

    const float scale  = 0.125f;                 // 1/sqrt(DK)
    const int nChunks  = (qs + 1) * 4;           // 32-key chunks this block needs

    // prologue: stage chunk 0 into buffer 0
#if USE_TDM
    if (w == 0) {
        tdm_load_2d(lds_off(&Ks[0][0][0]), Krow, 64, 32, 64);
        tdm_load_2d(lds_off(&Vs[0][0][0]), Vrow, 32, 64, Sc);
    }
#else
    cp16_g2l(Krow + (size_t)kR * DKc + kC, &Ks[0][kR][kC]);
    cp16_g2l(Vrow + (size_t)vR * Sc + vC,  &Vs[0][vR][vC]);
#endif

    for (int c = 0; c < nChunks; ++c) {
        const int buf = c & 1;
#if USE_TDM
        if (w == 0) __builtin_amdgcn_s_wait_tensorcnt(0);
#elif USE_ASYNC
        __builtin_amdgcn_s_wait_asynccnt(0);
#endif
        __syncthreads();                         // buf ready for all waves
        if (c + 1 < nChunks) {                   // overlap next copy w/ compute
            const int kn = (c + 1) * 32;
#if USE_TDM
            if (w == 0) {
                tdm_load_2d(lds_off(&Ks[buf ^ 1][0][0]),
                            Krow + (size_t)kn * DKc, 64, 32, 64);
                tdm_load_2d(lds_off(&Vs[buf ^ 1][0][0]),
                            Vrow + kn, 32, 64, Sc);
            }
#else
            cp16_g2l(Krow + (size_t)(kn + kR) * DKc + kC, &Ks[buf ^ 1][kR][kC]);
            cp16_g2l(Vrow + (size_t)vR * Sc + kn + vC,    &Vs[buf ^ 1][vR][vC]);
#endif
        }
        const int k0 = c * 32;
        if (k0 <= qbase + 15) {                  // wave-uniform causal skip
            // --- logits: two 16x16 tiles over DK=64 (2 WMMA k-steps each) ---
            v8f s0 = {}, s1 = {};
            s0 = wmma_bf16(aq0, load_frag_bf16(&Ks[buf][ 0][ 0], 64, lane), s0);
            s0 = wmma_bf16(aq1, load_frag_bf16(&Ks[buf][ 0][32], 64, lane), s0);
            s1 = wmma_bf16(aq0, load_frag_bf16(&Ks[buf][16][ 0], 64, lane), s1);
            s1 = wmma_bf16(aq1, load_frag_bf16(&Ks[buf][16][32], 64, lane), s1);

            // --- causal mask + online softmax (rows across 16-lane halves) ---
#pragma unroll
            for (int r = 0; r < 8; ++r) {
                const int m    = qbase + r + half * 8;
                const int key0 = k0 + nn;
                const int key1 = k0 + 16 + nn;
                float x0 = (key0 <= m) ? s0[r] * scale : -1.0e30f;
                float x1 = (key1 <= m) ? s1[r] * scale : -1.0e30f;
                float rowmax = fmaxf(x0, x1);
#pragma unroll
                for (int off = 1; off < 16; off <<= 1)
                    rowmax = fmaxf(rowmax, __shfl_xor(rowmax, off, 32));
                const float mnew  = fmaxf(mrun[r], rowmax);
                const float alpha = __expf(mrun[r] - mnew);
                const float p0 = __expf(x0 - mnew);
                const float p1 = __expf(x1 - mnew);
                float rsum = p0 + p1;
#pragma unroll
                for (int off = 1; off < 16; off <<= 1)
                    rsum += __shfl_xor(rsum, off, 32);
                lrun[r] = lrun[r] * alpha + rsum;
                mrun[r] = mnew;
                o0[r] *= alpha; o1[r] *= alpha; o2[r] *= alpha; o3[r] *= alpha;
                Pl[w][r + half * 8][nn]      = (__bf16)p0;
                Pl[w][r + half * 8][16 + nn] = (__bf16)p1;
            }
            // same-wave LDS RAW: DS ops are in-order; make dependency explicit
            asm volatile("s_wait_dscnt 0" ::: "memory");

            // --- P(16x32) x V(32x64) from LDS ---
            const v16bf pa = load_frag_bf16(&Pl[w][0][0], 32, lane);
            o0 = wmma_bf16(pa, load_frag_bf16(&Vs[buf][ 0][0], 32, lane), o0);
            o1 = wmma_bf16(pa, load_frag_bf16(&Vs[buf][16][0], 32, lane), o1);
            o2 = wmma_bf16(pa, load_frag_bf16(&Vs[buf][32][0], 32, lane), o2);
            o3 = wmma_bf16(pa, load_frag_bf16(&Vs[buf][48][0], 32, lane), o3);
        }
    }

    // --- normalize and store att tile as bf16 in [B,H,S,DK] flat order ---
#pragma unroll
    for (int r = 0; r < 8; ++r) {
        const float inv = 1.0f / lrun[r];
        const int s = qbase + r + half * 8;
        __bf16* dst = Att + ((size_t)bh * Sc + s) * DKc;
        dst[ 0 + nn] = (__bf16)(o0[r] * inv);
        dst[16 + nn] = (__bf16)(o1[r] * inv);
        dst[32 + nn] = (__bf16)(o2[r] * inv);
        dst[48 + nn] = (__bf16)(o3[r] * inv);
    }
}

// ---------------------------------------------------------------------------
// Kernel 3: output projection.  The reference's raw reshape of [B,H,S,dk]
// into [B,S,D] is exactly the flat memory order of Att, so Att is simply a
// row-major [B*S, D] matrix here.  Out = Att * Wo + bo (f32 output).
// ---------------------------------------------------------------------------
__global__ __launch_bounds__(256) void out_proj(
    const __bf16* __restrict__ Att, const __bf16* __restrict__ WtO,
    const float* __restrict__ bo, float* __restrict__ Out)
{
    const int lane = threadIdx.x & 31;
    const int w    = threadIdx.x >> 5;
    const int mt   = blockIdx.x;                 // 0..255
    const int nt   = blockIdx.y * 8 + w;         // 0..63

    const __bf16* Abase = Att + (size_t)(mt * 16) * Dm;
    const __bf16* Bbase = WtO + (size_t)(nt * 16) * Dm;

    v8f acc = {};
#pragma unroll 4
    for (int kc = 0; kc < Dm; kc += 32) {
        if (kc + 128 < Dm) {
            __builtin_prefetch(Abase + kc + 128 + (size_t)(lane & 15) * Dm, 0, 3);
            __builtin_prefetch(Bbase + kc + 128 + (size_t)(lane & 15) * Dm, 0, 3);
        }
        v16bf a = load_frag_bf16(Abase + kc, Dm, lane);
        v16bf b = load_frag_bf16(Bbase + kc, Dm, lane);
        acc = wmma_bf16(a, b, acc);
    }

    const int half = lane >> 4;
    const int n    = nt * 16 + (lane & 15);
    const float bb = bo[n];
#pragma unroll
    for (int r = 0; r < 8; ++r) {
        const int m = mt * 16 + r + half * 8;
        Out[(size_t)m * Dm + n] = acc[r] + bb;
    }
}

// ---------------------------------------------------------------------------
extern "C" void kernel_launch(void* const* d_in, const int* in_sizes, int n_in,
                              void* d_out, int out_size, void* d_ws, size_t ws_size,
                              hipStream_t stream) {
    const float* q   = (const float*)d_in[0];
    const float* k   = (const float*)d_in[1];
    const float* v   = (const float*)d_in[2];
    /* d_in[3] = causal mask: applied analytically in-kernel */
    const float* Wq  = (const float*)d_in[4];
    const float* bq  = (const float*)d_in[5];
    const float* Wk  = (const float*)d_in[6];
    const float* bk  = (const float*)d_in[7];
    const float* Wv  = (const float*)d_in[8];
    const float* bv  = (const float*)d_in[9];
    const float* Wo  = (const float*)d_in[10];
    const float* bo  = (const float*)d_in[11];
    float* out = (float*)d_out;

    // Workspace layout (bf16 elements): 4 transposed weights + Q,K,Vt,Att
    __bf16* ws  = (__bf16*)d_ws;
    __bf16* WtQ = ws;
    __bf16* WtK = WtQ + (size_t)Dm * Dm;
    __bf16* WtV = WtK + (size_t)Dm * Dm;
    __bf16* WtO = WtV + (size_t)Dm * Dm;
    __bf16* Qb  = WtO + (size_t)Dm * Dm;
    __bf16* Kb  = Qb  + (size_t)Bc * Sc * Dm;
    __bf16* Vt  = Kb  + (size_t)Bc * Sc * Dm;
    __bf16* Att = Vt  + (size_t)Bc * Sc * Dm;    // total 40 MB

    prep_w  <<<dim3((Dm * Dm) / 256, 4), 256, 0, stream>>>(Wq, Wk, Wv, Wo,
                                                           WtQ, WtK, WtV, WtO);
    qkv_proj<<<dim3(Bc * Sc / 16, Dm / 128, 3), 256, 0, stream>>>(
        q, k, v, WtQ, WtK, WtV, bq, bk, bv, Qb, Kb, Vt);
    attn    <<<dim3(Bc * Hc, Sc / 128), 256, 0, stream>>>(Qb, Kb, Vt, Att);
    out_proj<<<dim3(Bc * Sc / 16, Dm / 128), 256, 0, stream>>>(Att, WtO, bo, out);
}
